// ProbAttention_37555194036920
// MI455X (gfx1250) — matrix operand order repaired and checked by reading, not compiled, not run on hardware
//
#include <hip/hip_runtime.h>

// ---------------------------------------------------------------------------
// ProbSparse attention (Informer) for MI455X / gfx1250, wave32 + WMMA bf16.
// Shapes fixed by setup_inputs(): B=2, L=4096, D=512, H=8, d=64, topk=45.
// ---------------------------------------------------------------------------

typedef __bf16          v16bf    __attribute__((ext_vector_type(16)));
typedef float           v8f      __attribute__((ext_vector_type(8)));
typedef float           float4v  __attribute__((ext_vector_type(4)));
typedef unsigned int    uint4v   __attribute__((ext_vector_type(4)));
typedef unsigned int    uint8v   __attribute__((ext_vector_type(8)));
typedef unsigned short  ushort8  __attribute__((ext_vector_type(8)));

constexpr int NB  = 2;     // batch
constexpr int NL  = 4096;  // sequence length
constexpr int ND  = 512;   // d_model
constexpr int NH  = 8;     // heads
constexpr int NBH = NB * NH;
constexpr int NCC = 16;    // cumsum chunks
constexpr int NCH = NL / NCC;  // 256 rows per chunk
constexpr int TOP_STRIDE = 64; // padded top-k stride

__device__ __forceinline__ unsigned short bf16rne(float x) {
  unsigned u = __builtin_bit_cast(unsigned, x);
  unsigned r = u + 0x7FFFu + ((u >> 16) & 1u);
  return (unsigned short)(r >> 16);
}

// ---------------------------------------------------------------------------
// K0a: convert + transpose weights: W[k][n] fp32  ->  Wt[n][k] bf16
// ---------------------------------------------------------------------------
__global__ void w_transpose_bf16(const float* __restrict__ W0,
                                 const float* __restrict__ W1,
                                 const float* __restrict__ W2,
                                 unsigned short* __restrict__ T0,
                                 unsigned short* __restrict__ T1,
                                 unsigned short* __restrict__ T2) {
  const float* W = (blockIdx.y == 0) ? W0 : ((blockIdx.y == 1) ? W1 : W2);
  unsigned short* T = (blockIdx.y == 0) ? T0 : ((blockIdx.y == 1) ? T1 : T2);
  int gid = (int)blockIdx.x * 256 + (int)threadIdx.x;  // over ND*ND outputs
  int n = gid >> 9, k = gid & (ND - 1);
  T[gid] = bf16rne(W[k * ND + n]);
}

// ---------------------------------------------------------------------------
// K0b: bulk fp32 -> bf16 conversion (8 elements / thread, b128 in, b128 out)
// ---------------------------------------------------------------------------
__global__ void to_bf16_bulk(const float* __restrict__ X,
                             unsigned short* __restrict__ Y) {
  int gid = (int)blockIdx.x * 256 + (int)threadIdx.x;
  const float4v a = *(const float4v*)(X + (size_t)gid * 8);
  const float4v b = *(const float4v*)(X + (size_t)gid * 8 + 4);
  ushort8 o;
  #pragma unroll
  for (int i = 0; i < 4; ++i) {
    o[i]     = bf16rne(a[i]);
    o[4 + i] = bf16rne(b[i]);
  }
  *(ushort8*)(Y + (size_t)gid * 8) = o;
}

// ---------------------------------------------------------------------------
// K1: projection GEMM via WMMA bf16:  Out[b][h][l][dh] = X @ W + bias
// Block = 256 threads = 8 waves; block tile 128(M) x 64(N).
// Each wave owns a 32x32 patch: 2 A-fragments x 2 B-fragments -> 4 WMMAs per
// K-step (32B loaded per lane per WMMA; 4 independent accumulation chains).
// ---------------------------------------------------------------------------
__global__ void proj_gemm_wmma(const unsigned short* __restrict__ Xbf,
                               const unsigned short* __restrict__ Wt,
                               const float* __restrict__ bias,
                               float* __restrict__ Out) {
  const int lane = (int)threadIdx.x & 31;
  const int wave = (int)threadIdx.x >> 5;
  const int wm = wave & 3, wn = wave >> 2;
  const int mr = lane & 15, lh = lane >> 4;

  const int mbase = (int)blockIdx.x * 128 + wm * 32;
  const int n0    = (int)blockIdx.y * 64 + wn * 32;

  v8f acc00 = {}, acc01 = {}, acc10 = {}, acc11 = {};
  const unsigned short* a0row = Xbf + (size_t)(mbase + mr) * ND;
  const unsigned short* a1row = Xbf + (size_t)(mbase + 16 + mr) * ND;
  const unsigned short* b0row = Wt + (size_t)(n0 + mr) * ND;
  const unsigned short* b1row = Wt + (size_t)(n0 + 16 + mr) * ND;

  #pragma unroll 4
  for (int kk = 0; kk < ND; kk += 32) {
    // --- A fragments: lanes 0-15 hold
    //     K=[kk..kk+15]; lanes 16-31 shifted by +16.
    const int c0 = kk + lh * 16;
    uint8v av0, av1;
    {
      uint4v lo = *(const uint4v*)(a0row + c0);
      uint4v hi = *(const uint4v*)(a0row + c0 + 8);
      #pragma unroll
      for (int i = 0; i < 4; ++i) { av0[i] = lo[i]; av0[4 + i] = hi[i]; }
      lo = *(const uint4v*)(a1row + c0);
      hi = *(const uint4v*)(a1row + c0 + 8);
      #pragma unroll
      for (int i = 0; i < 4; ++i) { av1[i] = lo[i]; av1[4 + i] = hi[i]; }
    }
    v16bf a0 = __builtin_bit_cast(v16bf, av0);
    v16bf a1 = __builtin_bit_cast(v16bf, av1);

    // --- B fragments: lane holds 16 contiguous K of column n (Wt row-major)
    const int kb = kk + lh * 16;
    v16bf bA = __builtin_bit_cast(v16bf, *(const uint8v*)(b0row + kb));
    v16bf bB = __builtin_bit_cast(v16bf, *(const uint8v*)(b1row + kb));

    acc00 = __builtin_amdgcn_wmma_f32_16x16x32_bf16(false, a0, false, bA,
                                                    (short)0, acc00, false, false);
    acc01 = __builtin_amdgcn_wmma_f32_16x16x32_bf16(false, a0, false, bB,
                                                    (short)0, acc01, false, false);
    acc10 = __builtin_amdgcn_wmma_f32_16x16x32_bf16(false, a1, false, bA,
                                                    (short)0, acc10, false, false);
    acc11 = __builtin_amdgcn_wmma_f32_16x16x32_bf16(false, a1, false, bB,
                                                    (short)0, acc11, false, false);

    if (kk + 32 < ND) {  // -> global_prefetch_b8 for the next K-slab
      __builtin_prefetch(a0row + c0 + 32, 0, 3);
      __builtin_prefetch(a1row + c0 + 32, 0, 3);
      __builtin_prefetch(b0row + kb + 32, 0, 3);
      __builtin_prefetch(b1row + kb + 32, 0, 3);
    }
  }

  // --- epilogue: C/D layout: VGPR r -> M = r + 8*lh, N = lane&15
  const int ncolA = n0 + mr;
  const int ncolB = n0 + 16 + mr;
  const float biasA = bias[ncolA], biasB = bias[ncolB];
  const int hA = ncolA >> 6, dA = ncolA & 63;
  const int hB = ncolB >> 6, dB = ncolB & 63;
  #pragma unroll
  for (int s = 0; s < 2; ++s) {
    const int rbase = mbase + s * 16 + lh * 8;
    const v8f& aA = s ? acc10 : acc00;
    const v8f& aB = s ? acc11 : acc01;
    #pragma unroll
    for (int r = 0; r < 8; ++r) {
      int rowo = rbase + r;
      int b = rowo >> 12, l = rowo & (NL - 1);
      Out[(((size_t)(b * NH + hA) * NL + l) << 6) + dA] = aA[r] + biasA;
      Out[(((size_t)(b * NH + hB) * NL + l) << 6) + dB] = aB[r] + biasB;
    }
  }
}

// ---------------------------------------------------------------------------
// K2: sampled sparsity metric: m[bh][l] = max_s q.k_sample - sum_s q.k / L
// ---------------------------------------------------------------------------
__global__ void sample_score(const float* __restrict__ q,
                             const float* __restrict__ k,
                             const int* __restrict__ idxs,
                             float* __restrict__ marr, int SK) {
  int gid = (int)blockIdx.x * 256 + (int)threadIdx.x;  // over NBH*NL
  int bh = gid >> 12, l = gid & (NL - 1);
  const float* qr = q + ((size_t)bh * NL + l) * 64;
  float4v qv[16];
  #pragma unroll
  for (int i = 0; i < 16; ++i) qv[i] = *(const float4v*)(qr + i * 4);
  float mx = -__builtin_inff(), sm = 0.f;
  for (int s = 0; s < SK; ++s) {
    int ks = idxs[l * SK + s];
    const float* kr = k + ((size_t)bh * NL + ks) * 64;
    float d = 0.f;
    #pragma unroll
    for (int i = 0; i < 16; ++i) {
      float4v kv = *(const float4v*)(kr + i * 4);
      d = fmaf(qv[i][0], kv[0], d); d = fmaf(qv[i][1], kv[1], d);
      d = fmaf(qv[i][2], kv[2], d); d = fmaf(qv[i][3], kv[3], d);
    }
    mx = fmaxf(mx, d);
    sm += d;
  }
  marr[gid] = mx - sm * (1.0f / (float)NL);
}

// ---------------------------------------------------------------------------
// K3: iterative top-k (argmax + mask) per (b,h); tie-break = lowest index
// ---------------------------------------------------------------------------
__global__ void topk_iter(const float* __restrict__ marr,
                          int* __restrict__ mtop, int ntop) {
  __shared__ float sm[NL];
  __shared__ float rv[256];
  __shared__ int   ri[256];
  int bh = (int)blockIdx.x, tid = (int)threadIdx.x;
  for (int i = tid; i < NL; i += 256) sm[i] = marr[bh * NL + i];
  __syncthreads();
  for (int it = 0; it < ntop; ++it) {
    float bv = -__builtin_inff(); int bi = 0;
    for (int i = tid; i < NL; i += 256) {
      float v = sm[i];
      if (v > bv) { bv = v; bi = i; }
    }
    rv[tid] = bv; ri[tid] = bi;
    __syncthreads();
    for (int off = 128; off > 0; off >>= 1) {
      if (tid < off) {
        float ov = rv[tid + off]; int oi = ri[tid + off];
        if (ov > rv[tid] || (ov == rv[tid] && oi < ri[tid])) {
          rv[tid] = ov; ri[tid] = oi;
        }
      }
      __syncthreads();
    }
    if (tid == 0) { mtop[bh * TOP_STRIDE + it] = ri[0]; sm[ri[0]] = -__builtin_inff(); }
    __syncthreads();
  }
}

// ---------------------------------------------------------------------------
// K4a/b/c: cumsum of v over L into d_out (final (b,l,h*64+d) layout)
// ---------------------------------------------------------------------------
__global__ void vcumsum_chunk(const float* __restrict__ v,
                              float* __restrict__ out,
                              float* __restrict__ csum) {
  int bh = (int)blockIdx.x >> 4, c = (int)blockIdx.x & 15;
  int b = bh >> 3, h = bh & 7, dd = (int)threadIdx.x;
  const float* vb = v + ((size_t)bh * NL + c * NCH) * 64 + dd;
  float run = 0.f;
  for (int i = 0; i < NCH; ++i) {
    run += vb[(size_t)i * 64];
    out[((size_t)(b * NL + c * NCH + i) << 9) + h * 64 + dd] = run;
  }
  csum[(bh * NCC + c) * 64 + dd] = run;
}

__global__ void vcumsum_scan(float* __restrict__ csum) {
  int bh = (int)blockIdx.x, dd = (int)threadIdx.x;
  float ex = 0.f;
  for (int c = 0; c < NCC; ++c) {
    float t = csum[(bh * NCC + c) * 64 + dd];
    csum[(bh * NCC + c) * 64 + dd] = ex;
    ex += t;
  }
}

__global__ void vcumsum_add(float* __restrict__ out,
                            const float* __restrict__ csum) {
  int bh = (int)blockIdx.x >> 4, c = (int)blockIdx.x & 15;
  int b = bh >> 3, h = bh & 7;
  int dd = (int)threadIdx.x & 63, r0 = (int)threadIdx.x >> 6;
  float off = csum[(bh * NCC + c) * 64 + dd];
  for (int i = r0; i < NCH; i += 4)
    out[((size_t)(b * NL + c * NCH + i) << 9) + h * 64 + dd] += off;
}

// ---------------------------------------------------------------------------
// K5: sparse attention for the ntop selected query rows.
// One block per (b,h); K/V (2MB) stay L2-resident across the 45 rows.
// ---------------------------------------------------------------------------
__global__ void sparse_attn(const float* __restrict__ q,
                            const float* __restrict__ k,
                            const float* __restrict__ v,
                            const int* __restrict__ mtop,
                            float* __restrict__ out, int ntop) {
  __shared__ float qs[64];
  __shared__ float sc[NL];
  __shared__ float red[256];
  __shared__ float acc4[256];
  const int bh = (int)blockIdx.x, tid = (int)threadIdx.x;
  const int b = bh >> 3, h = bh & 7;
  const float scale = 0.125f;  // 1/sqrt(64)
  const float NEG = -__builtin_inff();

  for (int u = 0; u < ntop; ++u) {
    const int l0 = mtop[bh * TOP_STRIDE + u];
    if (tid < 64) qs[tid] = q[((size_t)bh * NL + l0) * 64 + tid];
    __syncthreads();

    // scores (causal: keys 0..l0)
    float lmax = NEG;
    for (int j = tid; j < NL; j += 256) {
      float s = NEG;
      if (j <= l0) {
        const float* kr = k + ((size_t)bh * NL + j) * 64;
        float d = 0.f;
        #pragma unroll
        for (int i = 0; i < 64; i += 4) {
          float4v kv = *(const float4v*)(kr + i);
          d = fmaf(qs[i], kv[0], d);     d = fmaf(qs[i + 1], kv[1], d);
          d = fmaf(qs[i + 2], kv[2], d); d = fmaf(qs[i + 3], kv[3], d);
        }
        s = d * scale;
      }
      sc[j] = s;
      lmax = fmaxf(lmax, s);
    }
    red[tid] = lmax; __syncthreads();
    for (int off = 128; off > 0; off >>= 1) {
      if (tid < off) red[tid] = fmaxf(red[tid], red[tid + off]);
      __syncthreads();
    }
    const float mx = red[0];
    __syncthreads();

    float lsum = 0.f;
    for (int j = tid; j < NL; j += 256) {
      float e = __expf(sc[j] - mx);
      sc[j] = e;
      lsum += e;
    }
    red[tid] = lsum; __syncthreads();
    for (int off = 128; off > 0; off >>= 1) {
      if (tid < off) red[tid] += red[tid + off];
      __syncthreads();
    }
    const float inv = 1.0f / red[0];

    // attn @ v  (4 partials per d-component)
    const int dd = tid & 63, part = tid >> 6;
    float a = 0.f;
    for (int j = part; j <= l0; j += 4)
      a = fmaf(sc[j], v[(((size_t)bh * NL + j) << 6) + dd], a);
    acc4[tid] = a;
    __syncthreads();
    if (tid < 64) {
      float s = acc4[tid] + acc4[64 + tid] + acc4[128 + tid] + acc4[192 + tid];
      out[((size_t)(b * NL + l0) << 9) + h * 64 + tid] = s * inv;
    }
    __syncthreads();
  }
}

// ---------------------------------------------------------------------------
extern "C" void kernel_launch(void* const* d_in, const int* in_sizes, int n_in,
                              void* d_out, int out_size, void* d_ws, size_t ws_size,
                              hipStream_t stream) {
  const float* queries = (const float*)d_in[0];
  const float* keys    = (const float*)d_in[1];
  const float* values  = (const float*)d_in[2];
  const float* Wq      = (const float*)d_in[3];
  const float* bq      = (const float*)d_in[4];
  const float* Wk      = (const float*)d_in[5];
  const float* bk      = (const float*)d_in[6];
  const float* Wv      = (const float*)d_in[7];
  const float* bv      = (const float*)d_in[8];
  const int*   idxs    = (const int*)d_in[9];
  float* out = (float*)d_out;

  const int SK = in_sizes[9] / NL;  // sample_k == n_top (l_q == l_k) == 45

  // workspace carving (256B aligned)
  char* base = (char*)d_ws;
  size_t off = 0;
  auto alloc = [&](size_t bytes) -> void* {
    void* p = base + off;
    off += bytes;
    off = (off + 255) & ~(size_t)255;
    return p;
  };
  unsigned short* wtq = (unsigned short*)alloc((size_t)ND * ND * 2);
  unsigned short* wtk = (unsigned short*)alloc((size_t)ND * ND * 2);
  unsigned short* wtv = (unsigned short*)alloc((size_t)ND * ND * 2);
  unsigned short* xqb = (unsigned short*)alloc((size_t)NB * NL * ND * 2);
  unsigned short* xkb = (unsigned short*)alloc((size_t)NB * NL * ND * 2);
  unsigned short* xvb = (unsigned short*)alloc((size_t)NB * NL * ND * 2);
  float* qws  = (float*)alloc((size_t)NBH * NL * 64 * 4);
  float* kws  = (float*)alloc((size_t)NBH * NL * 64 * 4);
  float* vws  = (float*)alloc((size_t)NBH * NL * 64 * 4);
  float* marr = (float*)alloc((size_t)NBH * NL * 4);
  int*   mtop = (int*)alloc((size_t)NBH * TOP_STRIDE * 4);
  float* csum = (float*)alloc((size_t)NBH * NCC * 64 * 4);

  const dim3 b256(256);

  w_transpose_bf16<<<dim3(ND * ND / 256, 3), b256, 0, stream>>>(
      Wq, Wk, Wv, wtq, wtk, wtv);

  const int nconv = NB * NL * ND / (8 * 256);  // blocks per tensor
  to_bf16_bulk<<<dim3(nconv), b256, 0, stream>>>(queries, xqb);
  to_bf16_bulk<<<dim3(nconv), b256, 0, stream>>>(keys,    xkb);
  to_bf16_bulk<<<dim3(nconv), b256, 0, stream>>>(values,  xvb);

  const dim3 gg(NB * NL / 128, ND / 64);
  proj_gemm_wmma<<<gg, b256, 0, stream>>>(xqb, wtq, bq, qws);
  proj_gemm_wmma<<<gg, b256, 0, stream>>>(xkb, wtk, bk, kws);
  proj_gemm_wmma<<<gg, b256, 0, stream>>>(xvb, wtv, bv, vws);

  sample_score<<<dim3(NBH * NL / 256), b256, 0, stream>>>(qws, kws, idxs, marr, SK);
  topk_iter<<<dim3(NBH), b256, 0, stream>>>(marr, mtop, SK);

  vcumsum_chunk<<<dim3(NBH * NCC), dim3(64), 0, stream>>>(vws, out, csum);
  vcumsum_scan<<<dim3(NBH), dim3(64), 0, stream>>>(csum);
  vcumsum_add<<<dim3(NBH * NCC), b256, 0, stream>>>(out, csum);

  sparse_attn<<<dim3(NBH), b256, 0, stream>>>(qws, kws, vws, mtop, out, SK);
}
